// SingleAttention_34780645163309
// MI455X (gfx1250) — matrix operand order repaired and checked by reading, not compiled, not run on hardware
//
#include <hip/hip_runtime.h>

typedef __attribute__((ext_vector_type(16))) _Float16     v16h;
typedef __attribute__((ext_vector_type(8)))  float        v8f;
typedef __attribute__((ext_vector_type(8)))  unsigned int v8u;
typedef __attribute__((ext_vector_type(4)))  int          v4i;

#define DIN   512
#define DK    256
#define DV    256
#define BATCH 8
#define SEQ   2048
#define ROWS  (BATCH * SEQ)   // 16384

// Async global->LDS copy (CDNA5 GLOBAL_LOAD_ASYNC_TO_LDS_B128, ASYNCcnt).
// Probe-derived signature: (v4i addrspace(1)*, v4i addrspace(3)*, imm offset, imm cpol)
#define GPTR(p) ((__attribute__((address_space(1))) v4i*)(p))
#define LPTR(p) ((__attribute__((address_space(3))) v4i*)(p))
#if __has_builtin(__builtin_amdgcn_global_load_async_to_lds_b128)
#define HAVE_ASYNC_LDS 1
#endif

// ---- CDNA5 WMMA f16 fragment index helpers (wave32) ----
// A matrix 16x32 f16: lane holds row M=lane&15. Dword d of the 8-dword frag
// holds K = {2d (d<4) | 8+2d (d>=4)} + (lane>=16 ? 8 : 0), consecutive pair.
__device__ __forceinline__ int a_k0(int d, int lane) {
    return (d < 4 ? 2 * d : 8 + 2 * d) + ((lane & 16) ? 8 : 0);
}
// B matrix 32x16 f16: lane holds col N=lane&15. Dword d holds
// K = {2d (d<4) | 8+2d (d>=4)} + (lane>=16 ? 8 : 0), consecutive pair.
__device__ __forceinline__ int b_k0(int d, int lane) {
    return (d < 4 ? 2 * d : 8 + 2 * d) + ((lane & 16) ? 8 : 0);
}

__device__ __forceinline__ v8f wmma_f16(v8u a, v8u b, v8f c) {
    return __builtin_amdgcn_wmma_f32_16x16x32_f16(
        false, __builtin_bit_cast(v16h, a),
        false, __builtin_bit_cast(v16h, b),
        (short)0, c, false, false);
}

// =====================================================================
// Kernel 1: fused projection GEMM  Y = X @ W + b   (fp32 in, f16 out)
// blockIdx.z: 0 = Q (scaled by 1/sqrt(dk), row-major), 1 = K (row-major),
//             2 = V (stored transposed per batch: Vt[b][d][s])
// Block = 256 threads = 8 waves; tile = 128 rows x 64 cols; K-step 32.
// (fp32->f16 conversion happens here, so this stage keeps the register
//  staging path; the pure-f16 attention stage uses async-to-LDS.)
// =====================================================================
__global__ __launch_bounds__(256)
void proj_kernel(const float* __restrict__ Xq, const float* __restrict__ Xk,
                 const float* __restrict__ Xv,
                 const float* __restrict__ Wq, const float* __restrict__ bq,
                 const float* __restrict__ Wk, const float* __restrict__ bk,
                 const float* __restrict__ Wv, const float* __restrict__ bv,
                 _Float16* __restrict__ Qh, _Float16* __restrict__ Kh,
                 _Float16* __restrict__ Vt)
{
    __shared__ _Float16 lds_x[128][40];   // 32 K + 8 pad
    __shared__ _Float16 lds_wt[64][40];   // W tile transposed: [n][k]

    const int tid  = threadIdx.x;
    const int lane = tid & 31;
    const int wid  = tid >> 5;
    const int bx   = blockIdx.x;          // row tile (128 rows)
    const int by   = blockIdx.y;          // col tile (64 cols)
    const int bz   = blockIdx.z;

    const float* X    = (bz == 0) ? Xq : (bz == 1) ? Xk : Xv;
    const float* W    = (bz == 0) ? Wq : (bz == 1) ? Wk : Wv;
    const float* bias = (bz == 0) ? bq : (bz == 1) ? bk : bv;
    const float scale = (bz == 0) ? 0.0625f : 1.0f;   // 1/sqrt(256) folded into Q

    const int row0 = bx * 128;
    const int n0   = by * 64;
    const int M    = lane & 15;
    const int hi   = (lane & 16) ? 8 : 0;

    v8f acc[4];
    #pragma unroll
    for (int ct = 0; ct < 4; ++ct) acc[ct] = (v8f){0.f,0.f,0.f,0.f,0.f,0.f,0.f,0.f};

    for (int k0 = 0; k0 < DIN; k0 += 32) {
        // stage X tile [128][32] fp32 -> f16 LDS
        #pragma unroll
        for (int i = 0; i < 4; ++i) {
            int c  = tid + i * 256;
            int r  = c >> 3;
            int c4 = (c & 7) * 4;
            const float4 x4 = *(const float4*)&X[(size_t)(row0 + r) * DIN + k0 + c4];
            lds_x[r][c4 + 0] = (_Float16)x4.x;
            lds_x[r][c4 + 1] = (_Float16)x4.y;
            lds_x[r][c4 + 2] = (_Float16)x4.z;
            lds_x[r][c4 + 3] = (_Float16)x4.w;
        }
        // stage W tile [32][64] fp32 -> f16 LDS, transposed
        #pragma unroll
        for (int i = 0; i < 2; ++i) {
            int c  = tid + i * 256;
            int r  = c >> 4;           // k within tile 0..31
            int cc = (c & 15) * 4;     // n within tile 0..60
            const float4 w4 = *(const float4*)&W[(size_t)(k0 + r) * DK + n0 + cc];
            lds_wt[cc + 0][r] = (_Float16)w4.x;
            lds_wt[cc + 1][r] = (_Float16)w4.y;
            lds_wt[cc + 2][r] = (_Float16)w4.z;
            lds_wt[cc + 3][r] = (_Float16)w4.w;
        }
        __syncthreads();

        v8u a;
        #pragma unroll
        for (int d = 0; d < 8; ++d)
            a[d] = *(const unsigned int*)&lds_x[wid * 16 + M][a_k0(d, lane)];
        #pragma unroll
        for (int ct = 0; ct < 4; ++ct) {
            v8u b;
            #pragma unroll
            for (int d = 0; d < 8; ++d)
                b[d] = *(const unsigned int*)&lds_wt[ct * 16 + M][b_k0(d, lane)];
            acc[ct] = wmma_f16(a, b, acc[ct]);
        }
        __syncthreads();
    }

    // epilogue: bias add, scale, f16 store
    #pragma unroll
    for (int ct = 0; ct < 4; ++ct) {
        const int col = n0 + ct * 16 + M;
        const float bv_ = bias[col];
        #pragma unroll
        for (int r = 0; r < 8; ++r) {
            const int rl   = wid * 16 + r + hi;
            const int grow = row0 + rl;
            _Float16 h = (_Float16)((acc[ct][r] + bv_) * scale);
            if (bz == 2) {
                int bb = grow >> 11, s = grow & 2047;
                Vt[((size_t)bb * DV + col) * SEQ + s] = h;   // transposed V
            } else if (bz == 1) {
                Kh[(size_t)grow * DK + col] = h;
            } else {
                Qh[(size_t)grow * DK + col] = h;
            }
        }
    }
}

// =====================================================================
// Kernel 2: flash attention. Block = 256 threads (8 waves), 128 query
// rows per block (16 per wave), key blocks of 32 streamed through LDS
// via the CDNA5 async global->LDS engine when available.
// =====================================================================
__global__ __launch_bounds__(256)
void attn_kernel(const _Float16* __restrict__ Qh, const _Float16* __restrict__ Kh,
                 const _Float16* __restrict__ Vt, float* __restrict__ out)
{
    __shared__ _Float16 lds_k[32][264];       // key rows, row-major (+8 pad)
    __shared__ _Float16 lds_v[256][40];       // V channels x 32 keys (+8 pad)
    __shared__ _Float16 lds_p[8][16][40];     // per-wave P staging

    const int tid  = threadIdx.x;
    const int lane = tid & 31;
    const int wid  = tid >> 5;
    const int bq_  = blockIdx.x;              // 128-row query tile in batch
    const int b    = blockIdx.y;
    const int M    = lane & 15;
    const int hi   = (lane & 16) ? 8 : 0;

    // resident Q fragments for this wave's 16 rows (scale pre-folded)
    const int qrow = bq_ * 128 + wid * 16 + M;
    v8u qf[8];
    #pragma unroll
    for (int kb = 0; kb < 8; ++kb)
        #pragma unroll
        for (int d = 0; d < 8; ++d)
            qf[kb][d] = *(const unsigned int*)
                &Qh[((size_t)b * SEQ + qrow) * DK + kb * 32 + a_k0(d, lane)];

    v8f acc[16];
    #pragma unroll
    for (int t = 0; t < 16; ++t) acc[t] = (v8f){0.f,0.f,0.f,0.f,0.f,0.f,0.f,0.f};
    float m_run[8], l_run[8];
    #pragma unroll
    for (int r = 0; r < 8; ++r) { m_run[r] = -1e30f; l_run[r] = 0.f; }

    for (int key0 = 0; key0 < SEQ; key0 += 32) {
        __syncthreads();
#if defined(HAVE_ASYNC_LDS)
        // async DMA: K block [32][256] and V block [256][32] straight to LDS
        #pragma unroll
        for (int i = 0; i < 4; ++i) {
            int c  = tid + i * 256;
            int r  = c >> 5;
            int c8 = (c & 31) * 8;
            __builtin_amdgcn_global_load_async_to_lds_b128(
                GPTR(&Kh[((size_t)b * SEQ + key0 + r) * DK + c8]),
                LPTR(&lds_k[r][c8]), 0, 0);
        }
        #pragma unroll
        for (int i = 0; i < 4; ++i) {
            int c  = tid + i * 256;
            int d  = c >> 2;
            int c8 = (c & 3) * 8;
            __builtin_amdgcn_global_load_async_to_lds_b128(
                GPTR(&Vt[((size_t)b * DV + d) * SEQ + key0 + c8]),
                LPTR(&lds_v[d][c8]), 0, 0);
        }
        asm volatile("s_wait_asynccnt 0" ::: "memory");
#else
        // fallback: register-staged copies
        #pragma unroll
        for (int i = 0; i < 4; ++i) {
            int c  = tid + i * 256;
            int r  = c >> 5;
            int c8 = (c & 31) * 8;
            *(uint4*)&lds_k[r][c8] =
                *(const uint4*)&Kh[((size_t)b * SEQ + key0 + r) * DK + c8];
        }
        #pragma unroll
        for (int i = 0; i < 4; ++i) {
            int c  = tid + i * 256;
            int d  = c >> 2;
            int c8 = (c & 3) * 8;
            *(uint4*)&lds_v[d][c8] =
                *(const uint4*)&Vt[((size_t)b * DV + d) * SEQ + key0 + c8];
        }
#endif
        if (key0 + 32 < SEQ) {
            __builtin_prefetch(&Kh[((size_t)b * SEQ + key0 + 32 + (tid >> 3)) * DK], 0, 1);
            __builtin_prefetch(&Vt[((size_t)b * DV + tid) * SEQ + key0 + 32], 0, 1);
        }
        __syncthreads();

        // scores: two 16x16 tiles, K=256 -> 8 WMMA each
        v8f s0 = (v8f){0.f,0.f,0.f,0.f,0.f,0.f,0.f,0.f};
        v8f s1 = (v8f){0.f,0.f,0.f,0.f,0.f,0.f,0.f,0.f};
        #pragma unroll
        for (int kb = 0; kb < 8; ++kb) {
            v8u b0, b1;
            #pragma unroll
            for (int d = 0; d < 8; ++d) {
                int kk = kb * 32 + b_k0(d, lane);
                b0[d] = *(const unsigned int*)&lds_k[M][kk];
                b1[d] = *(const unsigned int*)&lds_k[16 + M][kk];
            }
            s0 = wmma_f16(qf[kb], b0, s0);
            s1 = wmma_f16(qf[kb], b1, s1);
        }

        // online softmax per query row (row lives across 16 lanes of a VGPR)
        #pragma unroll
        for (int r = 0; r < 8; ++r) {
            float v0 = s0[r], v1 = s1[r];
            float mx = fmaxf(v0, v1);
            mx = fmaxf(mx, __shfl_xor(mx, 1));
            mx = fmaxf(mx, __shfl_xor(mx, 2));
            mx = fmaxf(mx, __shfl_xor(mx, 4));
            mx = fmaxf(mx, __shfl_xor(mx, 8));
            float mnew  = fmaxf(m_run[r], mx);
            float alpha = __expf(m_run[r] - mnew);
            float p0 = __expf(v0 - mnew);
            float p1 = __expf(v1 - mnew);
            float rs = p0 + p1;
            rs += __shfl_xor(rs, 1);
            rs += __shfl_xor(rs, 2);
            rs += __shfl_xor(rs, 4);
            rs += __shfl_xor(rs, 8);
            l_run[r] = l_run[r] * alpha + rs;
            m_run[r] = mnew;
            #pragma unroll
            for (int t = 0; t < 16; ++t) acc[t][r] *= alpha;
            lds_p[wid][r + hi][M]      = (_Float16)p0;
            lds_p[wid][r + hi][16 + M] = (_Float16)p1;
        }
        asm volatile("s_wait_dscnt 0" ::: "memory");

        // re-shape P (C layout -> A fragment) via wave-private LDS
        v8u pf;
        #pragma unroll
        for (int d = 0; d < 8; ++d)
            pf[d] = *(const unsigned int*)&lds_p[wid][M][a_k0(d, lane)];

        // P(16x32) @ V(32x256): 16 WMMAs
        #pragma unroll
        for (int t = 0; t < 16; ++t) {
            v8u bf;
            #pragma unroll
            for (int d = 0; d < 8; ++d)
                bf[d] = *(const unsigned int*)&lds_v[t * 16 + M][b_k0(d, lane)];
            acc[t] = wmma_f16(pf, bf, acc[t]);
        }
    }

    // normalize and store fp32 output
    #pragma unroll
    for (int r = 0; r < 8; ++r) {
        float inv = 1.0f / l_run[r];
        int rl = bq_ * 128 + wid * 16 + r + hi;
        #pragma unroll
        for (int t = 0; t < 16; ++t) {
            int col = t * 16 + M;
            out[((size_t)b * SEQ + rl) * DV + col] = acc[t][r] * inv;
        }
    }
}

extern "C" void kernel_launch(void* const* d_in, const int* in_sizes, int n_in,
                              void* d_out, int out_size, void* d_ws, size_t ws_size,
                              hipStream_t stream) {
    const float* q_in = (const float*)d_in[0];
    const float* k_in = (const float*)d_in[1];
    const float* v_in = (const float*)d_in[2];
    const float* Wq   = (const float*)d_in[3];
    const float* bq   = (const float*)d_in[4];
    const float* Wk   = (const float*)d_in[5];
    const float* bk   = (const float*)d_in[6];
    const float* Wv   = (const float*)d_in[7];
    const float* bv   = (const float*)d_in[8];

    _Float16* Qh = (_Float16*)d_ws;                       // 8 MB
    _Float16* Kh = Qh + (size_t)ROWS * DK;                // 8 MB
    _Float16* Vt = Kh + (size_t)ROWS * DK;                // 8 MB (transposed)

    dim3 g1(ROWS / 128, DK / 64, 3);
    proj_kernel<<<g1, 256, 0, stream>>>(q_in, k_in, v_in,
                                        Wq, bq, Wk, bk, Wv, bv,
                                        Qh, Kh, Vt);

    dim3 g2(SEQ / 128, BATCH);
    attn_kernel<<<g2, 256, 0, stream>>>(Qh, Kh, Vt, (float*)d_out);
}
